// ViT_Module_84765474554007
// MI455X (gfx1250) — matrix operand order repaired and checked by reading, not compiled, not run on hardware
//
#include <hip/hip_runtime.h>

// ---------------------------------------------------------------------------
// ViT encoder block for MI455X (gfx1250): bf16 WMMA (V_WMMA_F32_16X16X32_BF16)
// for all GEMMs + attention, f32 accumulate, LDS-tiled, wave32.
// CDNA5 data movement: TDM (tensor_load_to_lds) for GEMM A tiles,
// GLOBAL_LOAD_ASYNC_TO_LDS_B128 for attention V tiles, global prefetch.
// ---------------------------------------------------------------------------

typedef __attribute__((ext_vector_type(16))) __bf16   v16bf;
typedef __attribute__((ext_vector_type(8)))  float    v8f;
typedef __attribute__((ext_vector_type(4)))  unsigned v4u;
typedef __attribute__((ext_vector_type(8)))  unsigned v8u;
typedef __bf16 bf16;

#define HB    768
#define NB    1024
#define BB    8
#define NHB   12
#define DB    64
#define MLPD  3072
#define MROWS (BB * NB)   // 8192
#define BHN   (BB * NHB)  // 96

// ---- bf16 <-> f32: native converts (v_cvt / v_fma_mix*_bf16 on gfx1250) ----
__device__ __forceinline__ bf16  f2bf(float f) { return (bf16)f; }
__device__ __forceinline__ float bf2f(bf16 b)  { return (float)b; }

// ---- CDNA5 async global->LDS copy (16 bytes per lane, ASYNCcnt) ------------
// the LDS address is taken from the vdst VGPR; generic LDS pointers carry the LDS byte
// offset in their low 32 bits, so truncation yields the VGPR LDS address.
__device__ __forceinline__ void async_copy_b128(void* lds, const void* gptr) {
  unsigned l = (unsigned)(uintptr_t)lds;
  asm volatile("global_load_async_to_lds_b128 %0, %1, off"
               :: "v"(l), "v"(gptr) : "memory");
}
__device__ __forceinline__ void wait_async0() {
  asm volatile("s_wait_asynccnt 0" ::: "memory");
}

// ---- CDNA5 Tensor Data Mover: one-shot 64x32 bf16 tile -> padded LDS -------
// D# per ISA §8.3/8.4. Tile rows of 64B (16 DWORDs, pad_interval=3) get 16B
// (4 DWORDs, pad_amount=3) of LDS padding appended -> LDS row stride 80B,
// matching sA[64][40]. One instruction per tile, tracked with TENSORcnt.
__device__ __forceinline__ void tdm_load_a_tile(const void* gptr, void* lds,
                                                unsigned kd_elems) {
  unsigned long long ga = (unsigned long long)(uintptr_t)gptr;
  v4u g0;
  g0[0] = 1u;                                                 // count=1, user
  g0[1] = (unsigned)(uintptr_t)lds;                           // lds_addr
  g0[2] = (unsigned)ga;                                       // global_addr[31:0]
  g0[3] = (unsigned)((ga >> 32) & 0x01FFFFFFu) | (2u << 30);  // ga[56:32], type=2
  v8u g1;
  g1[0] = (1u << 16)            // data_size = 1 -> 2 bytes
        | (1u << 20)            // pad_enable
        | (3u << 22)            // pad_interval: 16 DWORDs between pads
        | (3u << 25);           // pad_amount:   4 DWORDs of pad
  g1[1] = (kd_elems & 0xFFFFu) << 16;                 // tensor_dim0[15:0]
  g1[2] = (kd_elems >> 16) | ((MROWS & 0xFFFFu) << 16); // dim0[31:16], dim1[15:0]
  g1[3] = (32u << 16);                                // dim1[31:16]=0, tile_dim0=32
  g1[4] = 64u;                                        // tile_dim1=64, tile_dim2=0
  g1[5] = kd_elems;                                   // tensor_dim0_stride[31:0]
  g1[6] = 0u;                                         // stride[47:32], dim1_stride
  g1[7] = 0u;
  asm volatile("tensor_load_to_lds %0, %1" :: "s"(g0), "s"(g1) : "memory");
}
__device__ __forceinline__ void wait_tensor0() {
  __builtin_amdgcn_s_wait_tensorcnt(0);
}

// ---- WMMA fragment loaders (CDNA5 ISA §7.12.2 layouts, wave32) -------------
// A matrix 16x32 bf16: lane L holds row M=L&15; lanes<16 -> K in
// {kb..kb+7, kb+16..kb+23} with kb=0; lanes>=16 -> kb=8.
__device__ __forceinline__ v16bf load_a_frag(const bf16* row, int kbA) {
  v16bf f;
#pragma unroll
  for (int i = 0; i < 8; ++i) {
    f[i]     = row[kbA + i];
    f[i + 8] = row[kbA + 16 + i];
  }
  return f;
}
// B matrix 32x16 bf16: lane L holds column N=L&15; lanes<16 -> K in
// {kb..kb+7, kb+16..kb+23} with kb=0; lanes>=16 -> kb=8.
__device__ __forceinline__ v16bf load_b_frag(const bf16* col, int kbB) {
  v16bf f;
#pragma unroll
  for (int i = 0; i < 16; ++i) f[i] = col[kbB + (i & 7) + ((i >> 3) << 4)];
  return f;
}

__device__ __forceinline__ v8f wmma_bf16(v16bf a, v16bf b, v8f c) {
  // (neg_a, A, neg_b, B, c_mod, C, reuse_a, reuse_b)
  return __builtin_amdgcn_wmma_f32_16x16x32_bf16(false, a, false, b,
                                                 (short)0, c, false, false);
}

// ---------------------------------------------------------------------------
// LayerNorm (f32 in -> bf16 out), one row of 768 per workgroup (256 threads).
// ---------------------------------------------------------------------------
__global__ void ln_bf16_kernel(const float* __restrict__ x,
                               const float* __restrict__ g,
                               const float* __restrict__ bta,
                               bf16* __restrict__ out) {
  const int row = blockIdx.x;
  const float* xr = x + (size_t)row * HB;
  const int tid = threadIdx.x;
  float v[3];
  float s = 0.f, sq = 0.f;
#pragma unroll
  for (int i = 0; i < 3; ++i) {
    v[i] = xr[tid + i * 256];
    s += v[i]; sq += v[i] * v[i];
  }
#pragma unroll
  for (int o = 1; o < 32; o <<= 1) {
    s  += __shfl_xor(s, o, 32);
    sq += __shfl_xor(sq, o, 32);
  }
  __shared__ float rs_[8], rq_[8];
  const int wave = tid >> 5, lane = tid & 31;
  if (lane == 0) { rs_[wave] = s; rq_[wave] = sq; }
  __syncthreads();
  float ts = 0.f, tq = 0.f;
#pragma unroll
  for (int i = 0; i < 8; ++i) { ts += rs_[i]; tq += rq_[i]; }
  const float mu   = ts * (1.0f / HB);
  const float var  = tq * (1.0f / HB) - mu * mu;
  const float rstd = rsqrtf(var + 1e-5f);
  bf16* orow = out + (size_t)row * HB;
#pragma unroll
  for (int i = 0; i < 3; ++i) {
    int c = tid + i * 256;
    orow[c] = f2bf((v[i] - mu) * rstd * g[c] + bta[c]);
  }
}

// ---------------------------------------------------------------------------
// Tiled bf16-WMMA GEMM: C[M,N] = A(bf16 MxK) * B(f32 KxN, converted to bf16).
// Block = 128 threads (4 waves); 64x64 output tile; BK=32.
// A tile arrives via one TDM descriptor (wave 0); B tile converted in flight.
// ---------------------------------------------------------------------------
enum { EPI_QKV = 0, EPI_BIAS_RES_F32 = 1, EPI_BIAS_GELU_BF16 = 2 };

template <int EPI>
__global__ void gemm_wmma_kernel(const bf16* __restrict__ A,
                                 const float* __restrict__ Bw,
                                 int Kd, int Nn,
                                 const float* __restrict__ bias,
                                 const float* __restrict__ res,
                                 float* __restrict__ outf,
                                 bf16* __restrict__ outb) {
  __shared__ bf16 sA[64][40];  // [m][k], padded (filled by TDM)
  __shared__ bf16 sB[64][40];  // [n][k] (transposed on store), padded
  const int tid  = threadIdx.x;          // 128
  const int wave = tid >> 5, lane = tid & 31;
  const int rb = blockIdx.y * 64, cb = blockIdx.x * 64;
  const int lrow = lane & 15;
  const int kbA = (lane < 16) ? 0 : 8;
  const int kbB = (lane < 16) ? 0 : 8;

  v8f acc[4] = {};

  for (int kt = 0; kt < Kd; kt += 32) {
    __syncthreads();
    // A tile: 64x32 bf16 via Tensor Data Mover, one descriptor (wave 0 only)
    if (wave == 0) {
      tdm_load_a_tile(&A[(size_t)rb * Kd + kt], &sA[0][0], (unsigned)Kd);
    }
    // B tile: 32x64 f32 -> bf16, stored transposed as sB[n][k]
#pragma unroll
    for (int i = 0; i < 16; ++i) {
      int e = tid * 16 + i;
      int k = e >> 6, n = e & 63;
      sB[n][k] = f2bf(Bw[(size_t)(kt + k) * Nn + cb + n]);
    }
    if (kt + 32 < Kd) {  // gfx1250 global_prefetch_b8 of the next tiles
      __builtin_prefetch(&A[(size_t)(rb + (tid >> 1)) * Kd + kt + 32], 0, 1);
      __builtin_prefetch(&Bw[(size_t)(kt + 32 + (tid >> 2)) * Nn + cb], 0, 1);
    }
    if (wave == 0) wait_tensor0();
    __syncthreads();

    v16bf af = load_a_frag(&sA[wave * 16 + lrow][0], kbA);
#pragma unroll
    for (int nt = 0; nt < 4; ++nt) {
      v16bf bfr = load_b_frag(&sB[nt * 16 + lrow][0], kbB);
      acc[nt] = wmma_bf16(af, bfr, acc[nt]);
    }
  }

  // Epilogue. D layout: elem j -> M = j + (lane<16?0:8), N = lane&15.
  const int mrow0 = rb + wave * 16 + ((lane < 16) ? 0 : 8);
#pragma unroll
  for (int nt = 0; nt < 4; ++nt) {
#pragma unroll
    for (int j = 0; j < 8; ++j) {
      const int r = mrow0 + j;
      const int c = cb + nt * 16 + lrow;
      const float vvl = acc[nt][j];
      if constexpr (EPI == EPI_QKV) {
        // scatter into [3][B][NH][N][D] bf16 for attention
        int bidx = r >> 10, n = r & (NB - 1);
        int which = c / HB, rem = c - which * HB;
        int hh = rem >> 6, dd = rem & 63;
        size_t dst = ((((size_t)which * BB + bidx) * NHB + hh) * NB + n) * DB + dd;
        outb[dst] = f2bf(vvl);
      } else if constexpr (EPI == EPI_BIAS_RES_F32) {
        size_t idx = (size_t)r * Nn + c;
        outf[idx] = vvl + bias[c] + res[idx];
      } else {  // bias + exact GELU -> bf16
        float t  = vvl + bias[c];
        float gl = 0.5f * t * (1.0f + erff(t * 0.70710678118f));
        outb[(size_t)r * Nn + c] = f2bf(gl);
      }
    }
  }
}

// ---------------------------------------------------------------------------
// Attention: one (b,h) head, one 16-query-row tile per workgroup (4 waves).
// Pass 1: S = Q K^T (WMMA, B-frags read directly from K rows, contiguous in d)
// -> bf16 logits in LDS. Two-phase softmax stats. Pass 2: O = P V (WMMA),
// A-frags built on the fly as exp(s - max) * 1/sum; V tiles staged to LDS via
// async global->LDS b128 copies.
// ---------------------------------------------------------------------------
__global__ void attn_kernel(const bf16* __restrict__ qkv, bf16* __restrict__ o) {
  const int bh = blockIdx.y;
  const int qt = blockIdx.x;
  const size_t hd = (size_t)NB * DB;  // 65536
  const bf16* Q = qkv + (size_t)bh * hd + (size_t)qt * 16 * DB;
  const bf16* K = qkv + ((size_t)BHN + bh) * hd;
  const bf16* V = qkv + ((size_t)2 * BHN + bh) * hd;

  const int tid  = threadIdx.x;       // 128
  const int wave = tid >> 5, lane = tid & 31;
  const int lrow = lane & 15;
  const int kbA = (lane < 16) ? 0 : 8;
  const int kbB = (lane < 16) ? 0 : 8;

  __shared__ bf16 s_sc[16][NB + 8];   // scaled logits, padded (33 KB)
  __shared__ bf16 s_v[32][72];        // V chunk [k][d], row padded to 144B
  __shared__ float s_mx[16], s_is[16];

  // Q fragments for this 16-row tile (rows contiguous in d)
  const bf16* qrow = Q + (size_t)lrow * DB;
  const v16bf qa0 = load_a_frag(qrow, kbA);
  const v16bf qa1 = load_a_frag(qrow + 32, kbA);

  // ---- pass 1: scores -----------------------------------------------------
  for (int jt = wave; jt < NB / 16; jt += 4) {
    const bf16* krow = K + (size_t)(jt * 16 + lrow) * DB;  // B column = key row
    v16bf k0 = load_b_frag(krow, kbB);        // d in [0,32)
    v16bf k1 = load_b_frag(krow + 32, kbB);   // d in [32,64)
    v8f sacc = {};
    sacc = wmma_bf16(qa0, k0, sacc);
    sacc = wmma_bf16(qa1, k1, sacc);
    const int col = jt * 16 + lrow;
    const int mb = (lane < 16) ? 0 : 8;
#pragma unroll
    for (int j = 0; j < 8; ++j)
      s_sc[mb + j][col] = f2bf(sacc[j] * 0.125f);  // * D^-0.5
  }
  __syncthreads();

  // ---- softmax stats: 8 threads per row ----------------------------------
  {
    const int row = tid >> 3, sub = tid & 7;
    float mx = -1e30f;
    for (int c = sub; c < NB; c += 8) mx = fmaxf(mx, bf2f(s_sc[row][c]));
#pragma unroll
    for (int o_ = 1; o_ < 8; o_ <<= 1) mx = fmaxf(mx, __shfl_xor(mx, o_, 32));
    float sm = 0.f;
    for (int c = sub; c < NB; c += 8) sm += __expf(bf2f(s_sc[row][c]) - mx);
#pragma unroll
    for (int o_ = 1; o_ < 8; o_ <<= 1) sm += __shfl_xor(sm, o_, 32);
    if (sub == 0) { s_mx[row] = mx; s_is[row] = 1.0f / sm; }
  }
  __syncthreads();

  // ---- pass 2: O = P * V; wave owns 16-wide d-slice -----------------------
  const int d0 = wave * 16;
  v8f oacc = {};
  const float mxr = s_mx[lrow], isr = s_is[lrow];
  for (int kt = 0; kt < NB / 32; ++kt) {
    __syncthreads();
    // stage 32x64 V chunk via async global->LDS (2x16B per thread)
#pragma unroll
    for (int cch = 0; cch < 2; ++cch) {
      int e = tid * 16 + cch * 8;        // 0..2047, 8-elem aligned
      int kk = e >> 6, dd = e & 63;
      async_copy_b128(&s_v[kk][dd], V + (size_t)(kt * 32 + kk) * DB + dd);
    }
    wait_async0();
    __syncthreads();
    v16bf af;
    const bf16* srow = &s_sc[lrow][kt * 32];
#pragma unroll
    for (int i = 0; i < 8; ++i) {
      af[i]     = f2bf(__expf(bf2f(srow[kbA + i])      - mxr) * isr);
      af[i + 8] = f2bf(__expf(bf2f(srow[kbA + 16 + i]) - mxr) * isr);
    }
    v16bf bfv;  // B frag: column d = d0+lrow, k = kbB..kbB+7, kbB+16..kbB+23 (strided LDS)
#pragma unroll
    for (int i = 0; i < 16; ++i) bfv[i] = s_v[kbB + (i & 7) + ((i >> 3) << 4)][d0 + lrow];
    oacc = wmma_bf16(af, bfv, oacc);
  }

  // write O as [B*N, H] row-major bf16 (A-matrix for the output projection)
  const int b = bh / NHB, h = bh - b * NHB;
  const int mb = (lane < 16) ? 0 : 8;
#pragma unroll
  for (int j = 0; j < 8; ++j) {
    const int nrow = qt * 16 + mb + j;
    size_t idx = ((size_t)b * NB + nrow) * HB + h * DB + d0 + lrow;
    o[idx] = f2bf(oacc[j]);
  }
}

// ---------------------------------------------------------------------------
// Host-side orchestration
// ---------------------------------------------------------------------------
extern "C" void kernel_launch(void* const* d_in, const int* in_sizes, int n_in,
                              void* d_out, int out_size, void* d_ws, size_t ws_size,
                              hipStream_t stream) {
  (void)in_sizes; (void)n_in; (void)out_size; (void)ws_size;
  const float* x     = (const float*)d_in[0];
  const float* w_qkv = (const float*)d_in[1];
  const float* w_out = (const float*)d_in[2];
  const float* b_out = (const float*)d_in[3];
  const float* w1    = (const float*)d_in[4];
  const float* b1    = (const float*)d_in[5];
  const float* w2    = (const float*)d_in[6];
  const float* b2    = (const float*)d_in[7];
  const float* ln1_g = (const float*)d_in[8];
  const float* ln1_b = (const float*)d_in[9];
  const float* ln2_g = (const float*)d_in[10];
  const float* ln2_b = (const float*)d_in[11];
  float* outp = (float*)d_out;

  // Workspace layout (~101 MB):
  //   [0]            ln_ws   bf16 [8192,768]            (reused for LN1, LN2)
  //   [+12.58MB]     qkv_ws  bf16 [3][8][12][1024][64]  (reused as mlp1 bf16)
  //   [+62.91MB]     o_ws    bf16 [8192,768]
  //   [+75.50MB]     x1_ws   f32  [8192,768]
  char* ws = (char*)d_ws;
  const size_t LN_BYTES  = (size_t)MROWS * HB * 2;      // 12,582,912
  const size_t BIG_BYTES = (size_t)MROWS * MLPD * 2;    // 50,331,648
  const size_t O_BYTES   = (size_t)MROWS * HB * 2;
  bf16* ln_ws   = (bf16*)(ws);
  bf16* qkv_ws  = (bf16*)(ws + LN_BYTES);
  bf16* mlp1_ws = qkv_ws;  // reuse: qkv dead after attention
  bf16* o_ws    = (bf16*)(ws + LN_BYTES + BIG_BYTES);
  float* x1_ws  = (float*)(ws + LN_BYTES + BIG_BYTES + O_BYTES);

  const dim3 blk(128);

  // 1. ln1(x) -> bf16
  ln_bf16_kernel<<<MROWS, 256, 0, stream>>>(x, ln1_g, ln1_b, ln_ws);
  // 2. qkv = ln1 @ w_qkv, scattered to [3][B][NH][N][D] bf16
  gemm_wmma_kernel<EPI_QKV><<<dim3(3 * HB / 64, MROWS / 64), blk, 0, stream>>>(
      ln_ws, w_qkv, HB, 3 * HB, nullptr, nullptr, nullptr, qkv_ws);
  // 3. attention per (b,h), 16 query rows per workgroup
  attn_kernel<<<dim3(NB / 16, BHN), blk, 0, stream>>>(qkv_ws, o_ws);
  // 4. x1 = x + o @ w_out + b_out  (f32)
  gemm_wmma_kernel<EPI_BIAS_RES_F32><<<dim3(HB / 64, MROWS / 64), blk, 0, stream>>>(
      o_ws, w_out, HB, HB, b_out, x, x1_ws, nullptr);
  // 5. ln2(x1) -> bf16
  ln_bf16_kernel<<<MROWS, 256, 0, stream>>>(x1_ws, ln2_g, ln2_b, ln_ws);
  // 6. h = gelu(ln2 @ w1 + b1) -> bf16
  gemm_wmma_kernel<EPI_BIAS_GELU_BF16><<<dim3(MLPD / 64, MROWS / 64), blk, 0, stream>>>(
      ln_ws, w1, HB, MLPD, b1, nullptr, nullptr, mlp1_ws);
  // 7. out = x1 + h @ w2 + b2  (f32)
  gemm_wmma_kernel<EPI_BIAS_RES_F32><<<dim3(HB / 64, MROWS / 64), blk, 0, stream>>>(
      mlp1_ws, w2, MLPD, HB, b2, x1_ws, outp, nullptr);
}
